// LSTM_41412074668225
// MI455X (gfx1250) — matrix operand order repaired and checked by reading, not compiled, not run on hardware
//
#include <hip/hip_runtime.h>

// Problem constants (from reference): B=128, T=1024, D=64, H=512, G=4H=2048
#define B_  128
#define T_  1024
#define D_  64
#define H_  512
#define G_  2048

typedef __bf16 v8bf  __attribute__((ext_vector_type(8)));
typedef __bf16 v16bf __attribute__((ext_vector_type(16)));
typedef float  v8f   __attribute__((ext_vector_type(8)));

__device__ __forceinline__ unsigned short f2bf(float f) {
  unsigned u = __builtin_bit_cast(unsigned, f);
  u += 0x7FFFu + ((u >> 16) & 1u);          // round-to-nearest-even
  return (unsigned short)(u >> 16);
}
__device__ __forceinline__ float bf2f(unsigned short s) {
  unsigned u = ((unsigned)s) << 16;
  return __builtin_bit_cast(float, u);
}
__device__ __forceinline__ v16bf cat16(v8bf lo, v8bf hi) {
  return __builtin_shufflevector(lo, hi, 0,1,2,3,4,5,6,7,8,9,10,11,12,13,14,15);
}
__device__ __forceinline__ float sigm(float x) { return 1.0f / (1.0f + __expf(-x)); }
__device__ __forceinline__ float tanh_fast(float x) {
  x = fminf(fmaxf(x, -15.0f), 15.0f);
  float e = __expf(2.0f * x);
  return (e - 1.0f) / (e + 1.0f);
}
__device__ __forceinline__ v8f wmma_bf16(v16bf a, v16bf b, v8f c) {
  return __builtin_amdgcn_wmma_f32_16x16x32_bf16(false, a, false, b, (short)0, c, false, false);
}
// Opaque 0: defeats LICM on loads whose address adds this, WITHOUT destroying
// the pointer's address-space provenance (keeps global_load, not flat_load).
__device__ __forceinline__ unsigned opaque_zero() {
  unsigned z = 0;
  asm volatile("" : "+v"(z));
  return z;
}

// ---------------------------------------------------------------- bf16 pack
__global__ void lstm_pack_bf16_kernel(const float* __restrict__ in,
                                      unsigned short* __restrict__ out, int n) {
  int i = blockIdx.x * blockDim.x + threadIdx.x;
  if (i < n) out[i] = f2bf(in[i]);
}

// ----------------------------------------------- gate pre-activation GEMM
// XG[M=B*T][G] = bf16( A[M][K] * W[G][K]^T + b_ih + b_hh )
template <int K>
__global__ __launch_bounds__(256)
void lstm_gate_gemm_kernel(const unsigned short* __restrict__ A,
                           const unsigned short* __restrict__ W,
                           const float* __restrict__ b_ih,
                           const float* __restrict__ b_hh,
                           unsigned short* __restrict__ XG) {
  constexpr int KP = K + 8;
  __shared__ unsigned short a_lds[16 * KP];
  const int tid  = threadIdx.x;
  const int wv   = tid >> 5;
  const int lane = tid & 31;
  const int grp  = lane >> 4;     // half-wave group (0/1)
  const int nl   = lane & 15;     // N within tile / M row for A loads
  const int mt    = blockIdx.y;
  const int nbase = blockIdx.x * 128 + wv * 16;

  for (int idx = tid * 8; idx < 16 * K; idx += 256 * 8) {
    const int r = idx / K;
    const int c = idx % K;
    *(v8bf*)&a_lds[r * KP + c] = *(const v8bf*)&A[((size_t)mt * 16 + r) * K + c];
  }
  __syncthreads();

  const unsigned short* Wr   = W + (size_t)(nbase + nl) * K;
  const unsigned short* arow = &a_lds[nl * KP];
  auto ldA = [&](int ka) -> v16bf {
    return cat16(*(const v8bf*)&arow[ka + grp * 8],
                 *(const v8bf*)&arow[ka + 16 + grp * 8]);
  };
  auto ldB = [&](int ka) -> v16bf {
    return cat16(*(const v8bf*)&Wr[ka + grp * 8],
                 *(const v8bf*)&Wr[ka + 16 + grp * 8]);
  };

  v8f acc = {};
  v16bf bc = ldB(0);
  #pragma unroll
  for (int kb = 0; kb < K / 32; ++kb) {
    v16bf bn = (kb + 1 < K / 32) ? ldB((kb + 1) * 32) : bc;   // prefetch next B
    v16bf a  = ldA(kb * 32);
    acc = wmma_bf16(a, bc, acc);
    bc = bn;
  }

  const float bsum = b_ih[nbase + nl] + b_hh[nbase + nl];
  #pragma unroll
  for (int v = 0; v < 8; ++v) {
    const int m = v + grp * 8;   // C/D layout: lane -> (M=v+8*grp, N=lane%16)
    XG[((size_t)mt * 16 + m) * G_ + nbase + nl] = f2bf(acc[v] + bsum);
  }
}

// ------------------------------------------------- persistent LSTM recurrence
// Grid = 8 blocks (16 batch rows each), 1024 threads = 32 waves (fills a WGP).
// Wave w owns h columns [16w,16w+16) and the 4 matching gate tiles; the cell
// state tile c lives in registers across all T steps; h (16x512 bf16) is the
// shared WMMA A-operand in LDS; W_hh (2MB bf16) streams from L2 every step.
// An opaque zero offset is mixed into the weight address each timestep so
// LICM cannot hoist (and spill) the t-invariant weight loads, while keeping
// global_load_b128 codegen (not flat_load) so W traffic stays off DScnt/LDS.
__global__ __launch_bounds__(1024)
void lstm_rec_kernel(const unsigned short* __restrict__ xg,
                     const unsigned short* __restrict__ Whh,
                     unsigned short* __restrict__ seq_out,
                     float* __restrict__ h_last,
                     int store_seq) {
  constexpr int HPAD = H_ + 8;                 // 1040B row stride: conflict-free
  constexpr size_t GSTRIDE = (size_t)H_ * H_;  // rows between gate blocks
  __shared__ unsigned short h_lds[16 * HPAD];
  const int tid  = threadIdx.x;
  const int wv   = tid >> 5;
  const int lane = tid & 31;
  const int grp  = lane >> 4;
  const int nl   = lane & 15;
  const int b0   = blockIdx.x * 16;
  const int hcol = wv * 16 + nl;

  for (int i = tid; i < 16 * HPAD; i += 1024) h_lds[i] = 0;

  const unsigned short* Wl   = Whh + (size_t)hcol * H_;   // gate-i row; others at +g*GSTRIDE
  const unsigned short* hrow = &h_lds[nl * HPAD];

  auto ldA = [&](int ka) -> v16bf {
    return cat16(*(const v8bf*)&hrow[ka + grp * 8],
                 *(const v8bf*)&hrow[ka + 16 + grp * 8]);
  };
  auto ldB = [&](const unsigned short* Wt, int g, int ka) -> v16bf {
    const unsigned short* p = Wt + (size_t)g * GSTRIDE + ka + grp * 8;
    return cat16(*(const v8bf*)p, *(const v8bf*)(p + 16));
  };

  v8f c_st = {};
  __syncthreads();

  for (int t = 0; t < T_; ++t) {
    // Prefetch next timestep's gate pre-activations (global_prefetch_b8).
    if (t + 1 < T_) {
      #pragma unroll
      for (int v = 0; v < 8; ++v) {
        const int m = v + grp * 8;
        const unsigned short* p = &xg[((size_t)(b0 + m) * T_ + (t + 1)) * G_ + hcol];
        __builtin_prefetch(p,           0, 1);
        __builtin_prefetch(p + 1 * H_,  0, 1);
        __builtin_prefetch(p + 2 * H_,  0, 1);
        __builtin_prefetch(p + 3 * H_,  0, 1);
      }
    }

    // Loop-variant (but zero) offset: weight loads cannot be hoisted out of
    // the t-loop; address space stays global.
    const unsigned short* Wt = Wl + opaque_zero();

    v8f acc0 = {}, acc1 = {}, acc2 = {}, acc3 = {};
    // Software-pipelined GEMM: h(16x512) x Whh^T -> 4 gate tiles (16x16 each)
    v16bf a_c = ldA(0);
    v16bf b0c = ldB(Wt, 0, 0), b1c = ldB(Wt, 1, 0), b2c = ldB(Wt, 2, 0), b3c = ldB(Wt, 3, 0);
    #pragma unroll
    for (int kb = 0; kb < H_ / 32; ++kb) {
      v16bf a_n, b0n, b1n, b2n, b3n;
      if (kb + 1 < H_ / 32) {
        const int kn = (kb + 1) * 32;
        a_n = ldA(kn);
        b0n = ldB(Wt, 0, kn); b1n = ldB(Wt, 1, kn);
        b2n = ldB(Wt, 2, kn); b3n = ldB(Wt, 3, kn);
      } else {
        a_n = a_c; b0n = b0c; b1n = b1c; b2n = b2c; b3n = b3c;
      }
      acc0 = wmma_bf16(a_c, b0c, acc0);
      acc1 = wmma_bf16(a_c, b1c, acc1);
      acc2 = wmma_bf16(a_c, b2c, acc2);
      acc3 = wmma_bf16(a_c, b3c, acc3);
      a_c = a_n; b0c = b0n; b1c = b1n; b2c = b2n; b3c = b3n;
    }

    // LSTM epilogue: add precomputed input-gate preacts, activations, c/h update
    float hval[8];
    #pragma unroll
    for (int v = 0; v < 8; ++v) {
      const int m = v + grp * 8;
      const size_t row = ((size_t)(b0 + m) * T_ + t) * G_;
      const float xi = bf2f(xg[row + 0 * H_ + hcol]);
      const float xf = bf2f(xg[row + 1 * H_ + hcol]);
      const float xc = bf2f(xg[row + 2 * H_ + hcol]);
      const float xo = bf2f(xg[row + 3 * H_ + hcol]);
      const float iv = sigm(acc0[v] + xi);
      const float fv = sigm(acc1[v] + xf);
      const float gv = tanh_fast(acc2[v] + xc);
      const float ov = sigm(acc3[v] + xo);
      const float cv = fv * c_st[v] + iv * gv;
      c_st[v] = cv;
      hval[v] = ov * tanh_fast(cv);
    }
    __syncthreads();  // all waves done reading h(t) from LDS
    #pragma unroll
    for (int v = 0; v < 8; ++v) {
      const int m = v + grp * 8;
      const unsigned short hb = f2bf(hval[v]);
      h_lds[m * HPAD + hcol] = hb;
      if (store_seq) seq_out[((size_t)(b0 + m) * T_ + t) * H_ + hcol] = hb;
    }
    __syncthreads();  // h(t+1) visible to all waves
    if (h_last != nullptr && t == T_ - 1) {
      #pragma unroll
      for (int v = 0; v < 8; ++v) {
        const int m = v + grp * 8;
        h_last[(size_t)(b0 + m) * H_ + hcol] = hval[v];
      }
    }
  }
}

// ---------------------------------------------------------------- FC head
__global__ void lstm_fc_head_kernel(const float* __restrict__ hlast,
                                    const float* __restrict__ fc1w,
                                    const float* __restrict__ fc1b,
                                    const float* __restrict__ fc2w,
                                    const float* __restrict__ fc2b,
                                    float* __restrict__ y) {
  int b = blockIdx.x * blockDim.x + threadIdx.x;
  if (b >= B_) return;
  const float* h = hlast + (size_t)b * H_;
  float acc2 = fc2b[0];
  for (int j = 0; j < 32; ++j) {
    float s = fc1b[j];
    const float* w = fc1w + (size_t)j * H_;
    #pragma unroll 8
    for (int k = 0; k < H_; ++k) s = fmaf(h[k], w[k], s);
    acc2 = fmaf(s, fc2w[j], acc2);
  }
  y[b] = acc2;
}

// ---------------------------------------------------------------- launcher
extern "C" void kernel_launch(void* const* d_in, const int* in_sizes, int n_in,
                              void* d_out, int out_size, void* d_ws, size_t ws_size,
                              hipStream_t stream) {
  const float* x    = (const float*)d_in[0];
  const float* Wih0 = (const float*)d_in[1];
  const float* Whh0 = (const float*)d_in[2];
  const float* bih0 = (const float*)d_in[3];
  const float* bhh0 = (const float*)d_in[4];
  const float* Wih1 = (const float*)d_in[5];
  const float* Whh1 = (const float*)d_in[6];
  const float* bih1 = (const float*)d_in[7];
  const float* bhh1 = (const float*)d_in[8];
  const float* fc1w = (const float*)d_in[9];
  const float* fc1b = (const float*)d_in[10];
  const float* fc2w = (const float*)d_in[11];
  const float* fc2b = (const float*)d_in[12];
  (void)in_sizes; (void)n_in; (void)out_size; (void)ws_size;

  char* ws = (char*)d_ws;
  size_t off = 0;
  auto alloc = [&](size_t bytes) -> void* {
    void* p = ws + off;
    off += (bytes + 255) & ~(size_t)255;
    return p;
  };
  unsigned short* xg   = (unsigned short*)alloc((size_t)B_ * T_ * G_ * 2);  // 512 MiB
  unsigned short* seq0 = (unsigned short*)alloc((size_t)B_ * T_ * H_ * 2);  // 128 MiB
  unsigned short* xbf  = (unsigned short*)alloc((size_t)B_ * T_ * D_ * 2);  //  16 MiB
  unsigned short* wih0 = (unsigned short*)alloc((size_t)G_ * D_ * 2);
  unsigned short* whh0 = (unsigned short*)alloc((size_t)G_ * H_ * 2);
  unsigned short* wih1 = (unsigned short*)alloc((size_t)G_ * H_ * 2);
  unsigned short* whh1 = (unsigned short*)alloc((size_t)G_ * H_ * 2);
  float*          hlast = (float*)alloc((size_t)B_ * H_ * 4);

  auto pack = [&](const float* src, unsigned short* dst, int n) {
    lstm_pack_bf16_kernel<<<(n + 255) / 256, 256, 0, stream>>>(src, dst, n);
  };
  pack(x, xbf, B_ * T_ * D_);
  pack(Wih0, wih0, G_ * D_);
  pack(Whh0, whh0, G_ * H_);
  pack(Wih1, wih1, G_ * H_);
  pack(Whh1, whh1, G_ * H_);

  const dim3 gg(G_ / 128, (B_ * T_) / 16);   // (16, 8192)

  // Layer 0
  lstm_gate_gemm_kernel<D_><<<gg, 256, 0, stream>>>(xbf, wih0, bih0, bhh0, xg);
  lstm_rec_kernel<<<B_ / 16, 1024, 0, stream>>>(xg, whh0, seq0, nullptr, 1);
  // Layer 1 (xg buffer reused)
  lstm_gate_gemm_kernel<H_><<<gg, 256, 0, stream>>>(seq0, wih1, bih1, bhh1, xg);
  lstm_rec_kernel<<<B_ / 16, 1024, 0, stream>>>(xg, whh1, nullptr, hlast, 0);
  // Head
  lstm_fc_head_kernel<<<1, 128, 0, stream>>>(hlast, fc1w, fc1b, fc2w, fc2b, (float*)d_out);
}
